// Seq2seq_58377195487985
// MI455X (gfx1250) — compile-verified, matrix operand order checked by reading, not run on hardware
//
#include <hip/hip_runtime.h>

// ---------------------------------------------------------------------------
// Seq2seq GRU encoder/decoder for MI455X (gfx1250, wave32, WMMA).
//   H=1024, MAX_INGR=64, MAX_LEN=128, OUT_V=30000
// Pipeline:
//   1) prep:       embedding gathers -> bf16 X matrices, init h0, reset barrier
//   2) gemm_wmma:  gi = X @ Wih^T + bih   (bf16 WMMA, f32 accum, B streamed once)
//   3) gru:        persistent cooperative kernel, Whh in LDS, async h staging
//   4) gemm_wmma:  logits = decH @ Wout^T + bout  -> d_out (Wout streamed once)
//   5) logsoftmax: in-place on d_out
// ---------------------------------------------------------------------------

typedef __attribute__((ext_vector_type(16))) __bf16          v16bf;
typedef __attribute__((ext_vector_type(8)))  float           v8f;
typedef __attribute__((ext_vector_type(4)))  float           v4f;
typedef __attribute__((ext_vector_type(8)))  unsigned short  v8us;
typedef __attribute__((ext_vector_type(16))) unsigned short  v16us;
typedef __attribute__((ext_vector_type(8)))  unsigned int    v8ui;

#define H      1024
#define T_ENC  64
#define T_DEC  128
#define OUTV   30000
#define NWG    128     // persistent GRU workgroups
#define GTH    128     // threads per GRU workgroup (4 waves)
#define ROWS   24      // 3 gates * 8 h-outputs per WG

__device__ __forceinline__ unsigned short f2bf_rne(float f) {
  unsigned u = __builtin_bit_cast(unsigned, f);
  u += 0x7FFFu + ((u >> 16) & 1u);
  return (unsigned short)(u >> 16);
}
__device__ __forceinline__ float bf2f(unsigned short h) {
  unsigned u = ((unsigned)h) << 16;
  return __builtin_bit_cast(float, u);
}
__device__ __forceinline__ float sigm(float x) { return 1.0f / (1.0f + expf(-x)); }

// {hi16(hi), hi16(lo)} in one dword via v_perm_b32
__device__ __forceinline__ unsigned pack_bf2(float lo, float hi) {
  return __builtin_amdgcn_perm(__builtin_bit_cast(unsigned, hi),
                               __builtin_bit_cast(unsigned, lo), 0x07060302u);
}

// ---------------------------------------------------------------------------
// 1) prep: gathers + h0 + barrier reset (deterministic every launch)
// ---------------------------------------------------------------------------
__global__ __launch_bounds__(256) void prep_kernel(
    const int* __restrict__ ingr, const int* __restrict__ tgt,
    const float* __restrict__ encE, const float* __restrict__ decE,
    unsigned short* __restrict__ Xe, unsigned short* __restrict__ Xd,
    float* __restrict__ hA, int* __restrict__ bar) {
  int b = blockIdx.x, t = threadIdx.x;
  if (b < T_ENC) {
    int tok = ingr[b];
    const float* src = encE + (size_t)tok * H;
    for (int k = t; k < H; k += 256) Xe[b * H + k] = f2bf_rne(src[k]);
  } else if (b < T_ENC + T_DEC) {
    int d = b - T_ENC;
    int tok = (d == 0) ? 0 : tgt[d - 1];           // SOS then teacher forcing
    const float* src = decE + (size_t)tok * H;
    for (int k = t; k < H; k += 256) {
      float x = src[k];
      x = x > 0.0f ? x : 0.0f;                     // relu on decoder embeds
      Xd[d * H + k] = f2bf_rne(x);
    }
  } else {
    for (int k = t; k < H; k += 256) hA[k] = 0.0f; // h0 = 0
    if (t == 0) *bar = 0;                          // reset grid barrier
  }
}

// ---------------------------------------------------------------------------
// 2/4) C[MT*16,N] = A_bf16[MT*16,K] @ W_f32[N,K]^T + bias[N]
//   One wave owns one 16-wide N tile and ALL MT m-tiles: the streamed f32
//   weight operand is loaded+converted ONCE and reused for MT WMMAs, so W
//   moves through HBM exactly once (the roofline term). A re-reads hit L2.
//   Software pipelined: batch all A loads, double-buffer the B stream.
// ---------------------------------------------------------------------------
template <int MT>
__global__ __launch_bounds__(256) void gemm_wmma(
    const unsigned short* __restrict__ A, const float* __restrict__ W,
    const float* __restrict__ bias, float* __restrict__ C,
    int N, int K, int Ntiles) {
  int wave = threadIdx.x >> 5;
  int nt   = blockIdx.x * 8 + wave;
  if (nt >= Ntiles) return;                        // uniform per wave
  int lane = threadIdx.x & 31;

  int ncol = nt * 16 + (lane & 15);
  int ksel = (lane >> 4) * 8;                      // ISA A/B per-lane K split

  const unsigned short* pa = A + (size_t)(lane & 15) * K + ksel;
  const float*          pw = W + (size_t)ncol * K + ksel;
  const size_t mstride = (size_t)16 * K;           // one m-tile of A

  auto loadB = [&](int k0) -> v16bf {
    v4f w0 = *(const v4f*)(pw + k0);
    v4f w1 = *(const v4f*)(pw + k0 + 4);
    v4f w2 = *(const v4f*)(pw + k0 + 16);
    v4f w3 = *(const v4f*)(pw + k0 + 20);
    v8ui ub;
    ub[0] = pack_bf2(w0[0], w0[1]); ub[1] = pack_bf2(w0[2], w0[3]);
    ub[2] = pack_bf2(w1[0], w1[1]); ub[3] = pack_bf2(w1[2], w1[3]);
    ub[4] = pack_bf2(w2[0], w2[1]); ub[5] = pack_bf2(w2[2], w2[3]);
    ub[6] = pack_bf2(w3[0], w3[1]); ub[7] = pack_bf2(w3[2], w3[3]);
    return __builtin_bit_cast(v16bf, ub);
  };

  v8f acc[MT];
#pragma unroll
  for (int m = 0; m < MT; ++m) acc[m] = v8f{};

  v16bf bv = loadB(0);
  for (int k0 = 0; k0 < K; k0 += 32) {
    // batch ALL A-fragment loads into distinct registers (one long clause)
    v16bf av[MT];
#pragma unroll
    for (int m = 0; m < MT; ++m) {
      const unsigned short* pam = pa + m * mstride + k0;
      v8us a0 = *(const v8us*)(pam);
      v8us a1 = *(const v8us*)(pam + 16);
      v16us ua = __builtin_shufflevector(a0, a1, 0, 1, 2, 3, 4, 5, 6, 7,
                                         8, 9, 10, 11, 12, 13, 14, 15);
      av[m] = __builtin_bit_cast(v16bf, ua);
    }
    // keep the W HBM stream ahead of the matrix pipe
    v16bf bnext = bv;
    if (k0 + 32 < K) {
      __builtin_prefetch(pw + k0 + 160, 0, 1);     // global_prefetch_b8
      bnext = loadB(k0 + 32);
    }
#pragma unroll
    for (int m = 0; m < MT; ++m)
      acc[m] = __builtin_amdgcn_wmma_f32_16x16x32_bf16(
          false, av[m], false, bv, (short)0, acc[m], false, false);
    bv = bnext;
  }

  float bb = bias[ncol];
#pragma unroll
  for (int m = 0; m < MT; ++m) {
    int mbase = m * 16 + (lane >> 4) * 8;          // D layout: VGPR r -> M=r(+8)
#pragma unroll
    for (int r = 0; r < 8; ++r)
      C[(size_t)(mbase + r) * N + ncol] = acc[m][r] + bb;
  }
}

// ---------------------------------------------------------------------------
// 3) persistent GRU: 128 WGs x 128 threads; each WG owns 8 h-outputs,
//    keeps its 24x1024 Whh slice in LDS (bf16), grid-syncs every step.
//    h vector staged into LDS each step with async-DMA (ASYNCcnt path).
// ---------------------------------------------------------------------------
__device__ __forceinline__ void grid_sync(int* bar, int target) {
  __syncthreads();
  if (threadIdx.x == 0) {
    __threadfence();
    __hip_atomic_fetch_add(bar, 1, __ATOMIC_RELEASE, __HIP_MEMORY_SCOPE_AGENT);
    while (__hip_atomic_load(bar, __ATOMIC_ACQUIRE, __HIP_MEMORY_SCOPE_AGENT) < target)
      __builtin_amdgcn_s_sleep(1);
  }
  __syncthreads();
}

__global__ __launch_bounds__(GTH) void gru_kernel(
    const float* __restrict__ encWhh, const float* __restrict__ encBhh,
    const float* __restrict__ giE,
    const float* __restrict__ decWhh, const float* __restrict__ decBhh,
    const float* __restrict__ giD,
    float* __restrict__ hA, float* __restrict__ hB,
    unsigned short* __restrict__ decH, int* __restrict__ bar) {
  __shared__ unsigned short wlds[ROWS * H];        // 48 KB
  __shared__ float hlds[H];                        // 4 KB
  __shared__ float ghs[ROWS];

  int wg = blockIdx.x, t = threadIdx.x;
  int wave = t >> 5, lane = t & 31;
  // raw LDS byte offset of hlds (low 32 bits of the flat shared address)
  unsigned hlds_base = (unsigned)(size_t)&hlds[0];

  const float* Whh_[2] = {encWhh, decWhh};
  const float* bhh_[2] = {encBhh, decBhh};
  const float* gi_[2]  = {giE, giD};
  const int steps_[2]  = {T_ENC, T_DEC};
  float* bufs[2] = {hA, hB};

  int pp = 0, phase = 0;
  for (int ph = 0; ph < 2; ++ph) {
    // stage this WG's Whh slice into LDS as bf16 (read once per phase)
    const float* Whh = Whh_[ph];
    for (int idx = t; idx < ROWS * H; idx += GTH) {
      int row = idx >> 10, k = idx & (H - 1);
      int rg = (row >> 3) * H + wg * 8 + (row & 7);   // gate*H + global j
      wlds[idx] = f2bf_rne(Whh[(size_t)rg * H + k]);
    }
    __syncthreads();
    const float* bhh = bhh_[ph];
    const float* gi  = gi_[ph];

    for (int step = 0; step < steps_[ph]; ++step) {
      const float* src = bufs[pp];
      float* dst = bufs[pp ^ 1];
      // async-DMA h (4KB) global -> LDS: 2 x b128 per lane, wave-local wait
#pragma unroll
      for (int it = 0; it < H / (GTH * 4); ++it) {
        int k4 = (t + it * GTH) * 4;                  // float index, 16B chunks
        unsigned ldsa = hlds_base + (unsigned)k4 * 4u;
        const float* g = src + k4;
        asm volatile("global_load_async_to_lds_b128 %0, %1, off"
                     :: "v"(ldsa), "v"(g) : "memory");
      }
      asm volatile("s_wait_asynccnt 0x0" ::: "memory");
      __syncthreads();

      // 4 waves x 6 rows: length-1024 dots from LDS, shuffle-reduced
#pragma unroll
      for (int rr = 0; rr < 6; ++rr) {
        int row = wave * 6 + rr;
        const unsigned short* wr = &wlds[row << 10];
        float s = 0.0f;
#pragma unroll
        for (int i = 0; i < 32; ++i) {
          int k = lane + (i << 5);
          s += bf2f(wr[k]) * hlds[k];
        }
#pragma unroll
        for (int o = 16; o; o >>= 1) s += __shfl_down(s, o, 32);
        if (lane == 0) {
          int rg = (row >> 3) * H + wg * 8 + (row & 7);
          ghs[row] = s + bhh[rg];
        }
      }
      __syncthreads();

      if (t < 8) {
        int j = wg * 8 + t;
        const float* g = gi + (size_t)step * (3 * H);
        float r = sigm(g[j] + ghs[t]);
        float z = sigm(g[H + j] + ghs[8 + t]);
        float n = tanhf(g[2 * H + j] + r * ghs[16 + t]);
        float hn = (1.0f - z) * n + z * hlds[j];
        dst[j] = hn;
        if (ph == 1) decH[step * H + j] = f2bf_rne(hn);
      }
      ++phase;
      grid_sync(bar, phase * NWG);
      pp ^= 1;
    }
  }
}

// ---------------------------------------------------------------------------
// 5) in-place row log-softmax, one 1024-thread block per row, single pass I/O
// ---------------------------------------------------------------------------
__device__ __forceinline__ float breduce(float v, float* sh, bool ismax) {
  int lane = threadIdx.x & 31, w = threadIdx.x >> 5, nw = blockDim.x >> 5;
#pragma unroll
  for (int o = 16; o; o >>= 1) {
    float u = __shfl_down(v, o, 32);
    v = ismax ? fmaxf(v, u) : (v + u);
  }
  if (lane == 0) sh[w] = v;
  __syncthreads();
  if (w == 0) {
    v = (lane < nw) ? sh[lane] : (ismax ? -3.4e38f : 0.0f);
#pragma unroll
    for (int o = 16; o; o >>= 1) {
      float u = __shfl_down(v, o, 32);
      v = ismax ? fmaxf(v, u) : (v + u);
    }
    if (lane == 0) sh[0] = v;
  }
  __syncthreads();
  v = sh[0];
  __syncthreads();
  return v;
}

__global__ __launch_bounds__(1024) void logsoftmax_kernel(float* __restrict__ out, int N) {
  __shared__ float sh[32];
  int row = blockIdx.x, t = threadIdx.x;
  float* p = out + (size_t)row * N;
  float v[30];
  float mx = -3.4e38f;
#pragma unroll
  for (int i = 0; i < 30; ++i) {
    int idx = t + (i << 10);
    if (idx < N) { v[i] = p[idx]; mx = fmaxf(mx, v[i]); }
  }
  mx = breduce(mx, sh, true);
  float s = 0.0f;
#pragma unroll
  for (int i = 0; i < 30; ++i) {
    int idx = t + (i << 10);
    if (idx < N) s += expf(v[i] - mx);
  }
  s = breduce(s, sh, false);
  float ls = logf(s);
#pragma unroll
  for (int i = 0; i < 30; ++i) {
    int idx = t + (i << 10);
    if (idx < N) p[idx] = v[i] - mx - ls;
  }
}

// ---------------------------------------------------------------------------
extern "C" void kernel_launch(void* const* d_in, const int* in_sizes, int n_in,
                              void* d_out, int out_size, void* d_ws, size_t ws_size,
                              hipStream_t stream) {
  const int*   ingr   = (const int*)d_in[0];
  const int*   tgt    = (const int*)d_in[1];
  const float* encE   = (const float*)d_in[2];
  const float* encWih = (const float*)d_in[3];
  const float* encWhh = (const float*)d_in[4];
  const float* encBih = (const float*)d_in[5];
  const float* encBhh = (const float*)d_in[6];
  const float* decE   = (const float*)d_in[7];
  const float* decWih = (const float*)d_in[8];
  const float* decWhh = (const float*)d_in[9];
  const float* decBih = (const float*)d_in[10];
  const float* decBhh = (const float*)d_in[11];
  const float* Wout   = (const float*)d_in[12];
  const float* bout   = (const float*)d_in[13];
  float* out = (float*)d_out;

  char* ws = (char*)d_ws;
  size_t o = 0;
  int* bar = (int*)(ws + o);            o += 256;
  float* hA = (float*)(ws + o);         o += H * 4;
  float* hB = (float*)(ws + o);         o += H * 4;
  unsigned short* Xe = (unsigned short*)(ws + o); o += (size_t)T_ENC * H * 2;
  unsigned short* Xd = (unsigned short*)(ws + o); o += (size_t)T_DEC * H * 2;
  float* giE = (float*)(ws + o);        o += (size_t)T_ENC * 3 * H * 4;
  float* giD = (float*)(ws + o);        o += (size_t)T_DEC * 3 * H * 4;
  unsigned short* dH = (unsigned short*)(ws + o); o += (size_t)T_DEC * H * 2;

  prep_kernel<<<T_ENC + T_DEC + 1, 256, 0, stream>>>(ingr, tgt, encE, decE, Xe, Xd, hA, bar);
  // gi precompute: Ntiles = 3H/16 = 192 column tiles, 8 waves/block
  gemm_wmma<4><<<24, 256, 0, stream>>>(Xe, encWih, encBih, giE, 3 * H, H, 192);
  gemm_wmma<8><<<24, 256, 0, stream>>>(Xd, decWih, decBih, giD, 3 * H, H, 192);
  gru_kernel<<<NWG, GTH, 0, stream>>>(encWhh, encBhh, giE, decWhh, decBhh, giD, hA, hB, dH, bar);
  // logits: Ntiles = 30000/16 = 1875 column tiles
  gemm_wmma<8><<<235, 256, 0, stream>>>(dH, Wout, bout, out, OUTV, H, 1875);
  logsoftmax_kernel<<<T_DEC, 1024, 0, stream>>>(out, OUTV);
}
